// AttentionParams_52261162058366
// MI455X (gfx1250) — matrix operand order repaired and checked by reading, not compiled, not run on hardware
//
#include <hip/hip_runtime.h>
#include <hip/hip_bf16.h>

typedef __attribute__((ext_vector_type(16))) __bf16 bf16x16;
typedef __attribute__((ext_vector_type(8)))  float  f32x8;
typedef __attribute__((ext_vector_type(4)))  int    v4i;

#if defined(__has_builtin)
#if __has_builtin(__builtin_amdgcn_global_load_async_to_lds_b128)
#define HAVE_ASYNC_LDS 1
#endif
#if __has_builtin(__builtin_amdgcn_s_wait_asynccnt)
#define HAVE_WAIT_ASYNC 1
#endif
#endif

#ifdef HAVE_ASYNC_LDS
typedef __attribute__((address_space(1))) v4i gv4i;
typedef __attribute__((address_space(3))) v4i lv4i;
static __device__ __forceinline__ void async_cp16(const unsigned short* g,
                                                  unsigned short* l) {
    __builtin_amdgcn_global_load_async_to_lds_b128(
        (gv4i*)(const void*)g, (lv4i*)(void*)l, 0, 0);
}
#endif

template <int N>
static __device__ __forceinline__ void wait_async() {
#ifdef HAVE_WAIT_ASYNC
    __builtin_amdgcn_s_wait_asynccnt(N);
#elif defined(HAVE_ASYNC_LDS)
    asm volatile("s_wait_asynccnt %0" ::"i"(N) : "memory");
#endif
}

static __device__ __forceinline__ unsigned short f2bf(float f) {
    unsigned int u = __float_as_uint(f);
    u += 0x7FFFu + ((u >> 16) & 1u);      // round-to-nearest-even
    return (unsigned short)(u >> 16);
}

// ---------------------------------------------------------------------------
// f32 -> bf16 elementwise convert (vectorized 4/thread)
// ---------------------------------------------------------------------------
__global__ void cvt_f32_bf16(const float* __restrict__ in,
                             unsigned short* __restrict__ out, int n4) {
    int i = blockIdx.x * blockDim.x + threadIdx.x;
    if (i < n4) {
        float4 f = ((const float4*)in)[i];
        ushort4 o;
        o.x = f2bf(f.x); o.y = f2bf(f.y); o.z = f2bf(f.z); o.w = f2bf(f.w);
        ((ushort4*)out)[i] = o;
    }
}

// ---------------------------------------------------------------------------
// f32 [D,D] -> bf16 [D,D] transposed (LDS-tiled)
// ---------------------------------------------------------------------------
__global__ void transpose_f32_bf16(const float* __restrict__ in,
                                   unsigned short* __restrict__ out, int D) {
    __shared__ float tile[32][33];
    int bx = blockIdx.x * 32, by = blockIdx.y * 32;
#pragma unroll
    for (int r = 0; r < 4; ++r)
        tile[threadIdx.y + r * 8][threadIdx.x] =
            in[(size_t)(by + threadIdx.y + r * 8) * D + bx + threadIdx.x];
    __syncthreads();
#pragma unroll
    for (int r = 0; r < 4; ++r)
        out[(size_t)(bx + threadIdx.y + r * 8) * D + by + threadIdx.x] =
            f2bf(tile[threadIdx.x][threadIdx.y + r * 8]);
}

// ---------------------------------------------------------------------------
// NT GEMM: C[M,N] = A[M,K] * B[N,K]^T   (bf16 in, f32 accum via WMMA)
// Block tile 128(M) x 256(N), K-tile 32. 8 waves (2x4), wave tile 64x64.
// Double-buffered LDS filled with global_load_async_to_lds_b128 (ASYNCcnt);
// A-fragments double-buffered in VGPRs to hide LDS latency behind WMMA.
// ---------------------------------------------------------------------------
template <bool OUT_F32>
__global__ __launch_bounds__(256, 2) void gemm_bf16_nt(
    const unsigned short* __restrict__ A, long long sA, int lda,
    const unsigned short* __restrict__ B, long long sB, int ldb,
    void* __restrict__ C, long long sC, int ldc,
    int M, int N, int K, int transT) {
    __shared__ unsigned short lA[2][128 * 40];   // 2 x (128 rows x 32 k +8 pad)
    __shared__ unsigned short lB[2][256 * 40];

    const int tid  = threadIdx.x;
    const int lane = tid & 31;
    const int wid  = tid >> 5;
    const int wm   = wid & 1;     // wave row (0..1)
    const int wn   = wid >> 1;    // wave col (0..3)
    const int bM   = blockIdx.y * 128;
    const int bN   = blockIdx.x * 256;

    A += (long long)blockIdx.z * sA;
    B += (long long)blockIdx.z * sB;

    f32x8 Cacc[4][4];
#pragma unroll
    for (int i = 0; i < 4; ++i)
#pragma unroll
        for (int j = 0; j < 4; ++j)
#pragma unroll
            for (int r = 0; r < 8; ++r) Cacc[i][j][r] = 0.0f;

    // staging: 16B chunks (8 bf16). A: 512 chunks, B: 1024 chunks, 256 thr.
    const unsigned short* pa[2];
    int aOff[2];
#pragma unroll
    for (int t = 0; t < 2; ++t) {
        int c = tid + t * 256;
        int row = c >> 2, cc = c & 3;
        pa[t]   = A + (size_t)(bM + row) * lda + cc * 8;
        aOff[t] = row * 40 + cc * 8;
    }
    const unsigned short* pb[4];
    int bOff[4];
#pragma unroll
    for (int t = 0; t < 4; ++t) {
        int c = tid + t * 256;
        int row = c >> 2, cc = c & 3;
        pb[t]   = B + (size_t)(bN + row) * ldb + cc * 8;
        bOff[t] = row * 40 + cc * 8;
    }

    const int arow = lane & 15;
    const int ahk  = (lane >> 4) * 8;   // A frag: chunks at +0/+16 (lo) or +8/+24 (hi)
    const int bhk  = (lane >> 4) * 8;   // B frag: chunks at +0/+16 (lo) or +8/+24 (hi)
    const int nIter = K >> 5;

#ifdef HAVE_ASYNC_LDS
    // prologue: async-fill buffer 0 with tile 0
#pragma unroll
    for (int t = 0; t < 2; ++t) async_cp16(pa[t], &lA[0][aOff[t]]);
#pragma unroll
    for (int t = 0; t < 4; ++t) async_cp16(pb[t], &lB[0][bOff[t]]);
#endif

    for (int it = 0; it < nIter; ++it) {
        const int cur = it & 1;
#ifdef HAVE_ASYNC_LDS
        if (it + 1 < nIter) {   // async-fill other buffer with next tile
            const int nxt = cur ^ 1, kn = (it + 1) << 5;
#pragma unroll
            for (int t = 0; t < 2; ++t) async_cp16(pa[t] + kn, &lA[nxt][aOff[t]]);
#pragma unroll
            for (int t = 0; t < 4; ++t) async_cp16(pb[t] + kn, &lB[nxt][bOff[t]]);
            wait_async<6>();    // my 6 loads for `cur` have landed
        } else {
            wait_async<0>();
        }
        __syncthreads();        // everyone's `cur` buffer ready
#else
        // fallback: synchronous staging into double buffer
        {
            const int k0 = it << 5;
            uint4 ta[2], tb[4];
#pragma unroll
            for (int t = 0; t < 2; ++t) ta[t] = *(const uint4*)(pa[t] + k0);
#pragma unroll
            for (int t = 0; t < 4; ++t) tb[t] = *(const uint4*)(pb[t] + k0);
#pragma unroll
            for (int t = 0; t < 2; ++t) *(uint4*)&lA[cur][aOff[t]] = ta[t];
#pragma unroll
            for (int t = 0; t < 4; ++t) *(uint4*)&lB[cur][bOff[t]] = tb[t];
        }
        __syncthreads();
#endif
        // ---- compute: 16 WMMAs. 4 B frags live; A frags double-buffered
        //      so group i+1's ds_loads issue before group i's WMMAs. ----
        union Frag { bf16x16 v; uint4 q[2]; };
        Frag fb[4];
#pragma unroll
        for (int j = 0; j < 4; ++j) {
            int r = wn * 64 + j * 16 + arow;
            fb[j].q[0] = *(const uint4*)&lB[cur][r * 40 + bhk];
            fb[j].q[1] = *(const uint4*)&lB[cur][r * 40 + bhk + 16];
        }
        Frag fa[2];
        {
            int r0 = wm * 64 + arow;
            fa[0].q[0] = *(const uint4*)&lA[cur][r0 * 40 + ahk];
            fa[0].q[1] = *(const uint4*)&lA[cur][r0 * 40 + ahk + 16];
        }
#pragma unroll
        for (int i = 0; i < 4; ++i) {
            if (i < 3) {   // prefetch next A fragment into the other buffer
                int rn = wm * 64 + (i + 1) * 16 + arow;
                fa[(i + 1) & 1].q[0] = *(const uint4*)&lA[cur][rn * 40 + ahk];
                fa[(i + 1) & 1].q[1] = *(const uint4*)&lA[cur][rn * 40 + ahk + 16];
            }
#pragma unroll
            for (int j = 0; j < 4; ++j)
                Cacc[i][j] = __builtin_amdgcn_wmma_f32_16x16x32_bf16(
                    false, fa[i & 1].v, false, fb[j].v, (short)0, Cacc[i][j],
                    false, false);
        }
        __syncthreads();        // all waves done reading `cur` before refill
    }

    // Epilogue. C layout: VGPR r -> row (r + 8*(lane>=16)), col = lane%16.
    const int col = lane & 15;
    const int rh  = (lane >> 4) * 8;
#pragma unroll
    for (int i = 0; i < 4; ++i)
#pragma unroll
        for (int j = 0; j < 4; ++j) {
            int gm0 = bM + wm * 64 + i * 16 + rh;
            int gn  = bN + wn * 64 + j * 16 + col;
#pragma unroll
            for (int r = 0; r < 8; ++r) {
                int gm  = gm0 + r;
                float v = Cacc[i][j][r];
                if (OUT_F32) {
                    ((float*)C)[(long long)blockIdx.z * sC +
                                (long long)gm * ldc + gn] = v;
                } else if (transT) {  // store C^T per batch of transT rows
                    int b = gm / transT, tl = gm - b * transT;
                    ((unsigned short*)C)[(long long)b * N * transT +
                                         (long long)gn * transT + tl] = f2bf(v);
                } else {
                    ((unsigned short*)C)[(long long)blockIdx.z * sC +
                                         (long long)gm * ldc + gn] = f2bf(v);
                }
            }
        }
}

// ---------------------------------------------------------------------------
extern "C" void kernel_launch(void* const* d_in, const int* in_sizes, int n_in,
                              void* d_out, int out_size, void* d_ws,
                              size_t ws_size, hipStream_t stream) {
    (void)in_sizes; (void)n_in; (void)out_size; (void)ws_size;
    const float* x  = (const float*)d_in[0];
    const float* wq = (const float*)d_in[1];
    const float* wk = (const float*)d_in[2];
    const float* wv = (const float*)d_in[3];
    const float* wo = (const float*)d_in[4];
    float* out = (float*)d_out;

    const int       Bb = 4, T = 4096, D = 1024;
    const long long M  = (long long)Bb * T;      // 16384

    unsigned short* ws  = (unsigned short*)d_ws;
    size_t o = 0;
    unsigned short* xb  = ws + o; o += (size_t)M * D;      // bf16(x)
    unsigned short* wqT = ws + o; o += (size_t)D * D;
    unsigned short* wkT = ws + o; o += (size_t)D * D;
    unsigned short* wvT = ws + o; o += (size_t)D * D;
    unsigned short* woT = ws + o; o += (size_t)D * D;
    unsigned short* qb  = ws + o; o += (size_t)M * D;      // q (bf16)
    unsigned short* kb  = ws + o; o += (size_t)M * D;      // k (bf16)
    unsigned short* vtb = ws + o; o += (size_t)M * D;      // v^T per batch [D,T]
    unsigned short* sb  = ws + o; o += (size_t)Bb * T * T; // scores (bf16)
    unsigned short* ab  = qb;  // attn output aliases q (q dead after scores)

    // --- precision conversion -------------------------------------------
    int n4 = (int)((M * D) / 4);
    cvt_f32_bf16<<<(n4 + 255) / 256, 256, 0, stream>>>(x, xb, n4);
    dim3 tg(D / 32, D / 32), tb(32, 8);
    transpose_f32_bf16<<<tg, tb, 0, stream>>>(wq, wqT, D);
    transpose_f32_bf16<<<tg, tb, 0, stream>>>(wk, wkT, D);
    transpose_f32_bf16<<<tg, tb, 0, stream>>>(wv, wvT, D);
    transpose_f32_bf16<<<tg, tb, 0, stream>>>(wo, woT, D);

    dim3 blk(256);
    // Q = x @ Wq   -> qb [16384,1024]
    gemm_bf16_nt<false><<<dim3(D / 256, M / 128, 1), blk, 0, stream>>>(
        xb, 0, D, wqT, 0, D, qb, 0, D, (int)M, D, D, 0);
    // K = x @ Wk   -> kb
    gemm_bf16_nt<false><<<dim3(D / 256, M / 128, 1), blk, 0, stream>>>(
        xb, 0, D, wkT, 0, D, kb, 0, D, (int)M, D, D, 0);
    // V = x @ Wv, stored transposed per batch -> vtb [4][1024][4096]
    gemm_bf16_nt<false><<<dim3(D / 256, M / 128, 1), blk, 0, stream>>>(
        xb, 0, D, wvT, 0, D, vtb, 0, 0, (int)M, D, D, T);
    // scores[b] = Q[b] @ K[b]^T  -> sb [4][4096][4096]
    gemm_bf16_nt<false><<<dim3(T / 256, T / 128, Bb), blk, 0, stream>>>(
        qb, (long long)T * D, D, kb, (long long)T * D, D,
        sb, (long long)T * T, T, T, T, D, 0);
    // attn[b] = scores[b] @ (V^T[b])^T -> ab [4][4096][1024]
    gemm_bf16_nt<false><<<dim3(D / 256, T / 128, Bb), blk, 0, stream>>>(
        sb, (long long)T * T, T, vtb, (long long)D * T, T,
        ab, (long long)T * D, D, T, D, T, 0);
    // out = attn @ Wo  -> f32 d_out [16384,1024]
    gemm_bf16_nt<true><<<dim3(D / 256, M / 128, 1), blk, 0, stream>>>(
        ab, 0, D, woT, 0, D, out, 0, D, (int)M, D, D, 0);
}
